// NoideAwareFeatureExtractor_86775519248851
// MI455X (gfx1250) — matrix operand order repaired and checked by reading, not compiled, not run on hardware
//
#include <hip/hip_runtime.h>
#include <hip/hip_bf16.h>
#include <cstdint>
#include <cstddef>

// ---------------------------------------------------------------------------
// Types for CDNA5 WMMA (wave32): v_wmma_f32_16x16x32_bf16
// ---------------------------------------------------------------------------
typedef __bf16          v16bf __attribute__((ext_vector_type(16)));
typedef unsigned short  v16us __attribute__((ext_vector_type(16)));
typedef float           v8f   __attribute__((ext_vector_type(8)));
typedef unsigned int    v4u   __attribute__((ext_vector_type(4)));
typedef int             v4i   __attribute__((ext_vector_type(4)));
typedef int             v8i   __attribute__((ext_vector_type(8)));

#if defined(__has_builtin)
#if __has_builtin(__builtin_amdgcn_tensor_load_to_lds) && \
    __has_builtin(__builtin_amdgcn_s_wait_tensorcnt)
#define ATHENA_HAS_TDM 1
#endif
#endif
#ifndef ATHENA_HAS_TDM
#define ATHENA_HAS_TDM 0
#endif

static __device__ __forceinline__ unsigned short f2bf(float f) {
  // round-to-nearest-even f32 -> bf16
  unsigned int u = __builtin_bit_cast(unsigned int, f);
  u += 0x7FFFu + ((u >> 16) & 1u);
  return (unsigned short)(u >> 16);
}

// A-fragment (16x32, 16-bit) layout: lane half = L>>4
//   element e -> K = e + 16*half
static __device__ __forceinline__ int frag_k_a(int e, int half) {
  return e + (half << 4);
}
// B-fragment (32x16, 16-bit): lanes 0-15 hold K=0..15, lanes 16-31 hold K=16..31
static __device__ __forceinline__ int frag_k_b(int e, int half) {
  return e + (half << 4);
}

#if ATHENA_HAS_TDM
// ---------------------------------------------------------------------------
// Tensor Data Mover: DMA a 2-D bf16 tile (rows x dim0, row stride stride0
// elements) from global memory into LDS at lds_addr. D# layout per
// CDNA5 ISA (group0: count/lds/global/type, group1:
// data_size, tensor dims, tile dims, stride0). Groups 2/3 disabled (2-D).
// ---------------------------------------------------------------------------
static __device__ __forceinline__ void tdm_load_tile_2d(
    unsigned lds_addr, const void* gptr,
    unsigned dim0, unsigned rows, unsigned stride0)
{
  unsigned long long ga = (unsigned long long)(uintptr_t)gptr;
  v4u g0;
  g0[0] = 1u;                                        // count=1, user mode
  g0[1] = lds_addr;                                  // LDS byte address
  g0[2] = (unsigned)(ga & 0xFFFFFFFFu);              // global_addr[31:0]
  g0[3] = (unsigned)((ga >> 32) & 0x01FFFFFFu)       // global_addr[56:32]
        | (2u << 30);                                // type=2 ("image")
  v8i g1;
  g1[0] = (int)(1u << 16);                           // wg_mask=0, data_size=2B
  g1[1] = (int)((dim0 & 0xFFFFu) << 16);             // tensor_dim0[15:0]
  g1[2] = (int)((dim0 >> 16) | ((rows & 0xFFFFu) << 16));  // dim0 hi | dim1 lo
  g1[3] = (int)((rows >> 16) | ((dim0 & 0xFFFFu) << 16));  // dim1 hi | tile_dim0
  g1[4] = (int)(rows & 0xFFFFu);                     // tile_dim1=rows, tile_dim2=0
  g1[5] = (int)stride0;                              // tensor_dim0_stride[31:0]
  g1[6] = 0;                                         // stride0 hi, stride1 lo
  g1[7] = 0;                                         // stride1 hi
  v4i gz = {0, 0, 0, 0};
#if __clang_major__ >= 23
  v8i gz8 = {0, 0, 0, 0, 0, 0, 0, 0};
  __builtin_amdgcn_tensor_load_to_lds(g0, g1, gz, gz, gz8, 0);
#else
  __builtin_amdgcn_tensor_load_to_lds(g0, g1, gz, gz, 0);
#endif
}
#endif  // ATHENA_HAS_TDM

// ---------------------------------------------------------------------------
// Generic bf16 WMMA GEMM: out(MxN) = act(A(MxK) @ W(NxK)^T + bias)
// One block = 8 waves sharing a 16-column weight tile staged in LDS by the
// Tensor Data Mover; each wave computes one 16x16 output tile.
// ---------------------------------------------------------------------------
template <int K, int N, int RELU>
__global__ __launch_bounds__(256) void gemm_wmma(
    const unsigned short* __restrict__ A,
    const unsigned short* __restrict__ W,
    const float* __restrict__ bias,
    unsigned short* __restrict__ out, int M)
{
  constexpr int tiles_n = N / 16;
  const int tn   = (int)(blockIdx.x % tiles_n);
  const int tm   = (int)(blockIdx.x / tiles_n) * 8 + (int)(threadIdx.x >> 5);
  const int lane = threadIdx.x & 31;
  const int half = lane >> 4;
  const int mr   = lane & 15;

  // Stage the 16 x K weight tile for this tn into LDS.
  __shared__ unsigned short sh_w[16 * K];
#if ATHENA_HAS_TDM
  if (threadIdx.x < 32) {  // wave 0 drives the DMA
    tdm_load_tile_2d((unsigned)(uintptr_t)&sh_w[0],
                     W + (size_t)tn * 16 * K, K, 16, K);
    __builtin_amdgcn_s_wait_tensorcnt(0);
  }
#else
  for (int i = threadIdx.x; i < 16 * K; i += blockDim.x)
    sh_w[i] = W[(size_t)tn * 16 * K + i];
#endif
  __syncthreads();

  if (tm * 16 >= M) return;  // wave-uniform: EXEC stays all-1s for WMMA

  const unsigned short* __restrict__ arow = A + (size_t)(tm * 16 + mr) * K;
  const int wbase = mr * K;

  v8f acc = {};
#pragma unroll
  for (int k0 = 0; k0 < K; k0 += 32) {
    if (k0 + 32 < K) __builtin_prefetch(arow + k0 + 32, 0, 1);  // global_prefetch_b8
    v16us au, bu;
#pragma unroll
    for (int e = 0; e < 16; ++e) {
      const int ka = k0 + frag_k_a(e, half);
      const int kb = k0 + frag_k_b(e, half);
      au[e] = (ka < K) ? arow[ka] : (unsigned short)0;
      bu[e] = (kb < K) ? sh_w[wbase + kb] : (unsigned short)0;
    }
    v16bf a = __builtin_bit_cast(v16bf, au);
    v16bf b = __builtin_bit_cast(v16bf, bu);
    acc = __builtin_amdgcn_wmma_f32_16x16x32_bf16(
        false, a, false, b, (short)0, acc, false, false);
  }

  const int  col = tn * 16 + mr;
  const float bv = bias[col];
#pragma unroll
  for (int r = 0; r < 8; ++r) {
    const int row = tm * 16 + r + 8 * half;
    float v = acc[r] + bv;
    if (RELU) v = fmaxf(v, 0.0f);
    out[(size_t)row * N + col] = f2bf(v);
  }
}

// ---------------------------------------------------------------------------
// Last layer fused: out(B,N) = max over points of relu(A @ W^T + b).
// Post-ReLU values are >= 0 so int atomicMax on the f32 bit pattern is exact.
// ---------------------------------------------------------------------------
template <int K, int N, int PTS>
__global__ __launch_bounds__(256) void gemm_wmma_maxout(
    const unsigned short* __restrict__ A,
    const unsigned short* __restrict__ W,
    const float* __restrict__ bias,
    float* __restrict__ out, int M)
{
  constexpr int tiles_n = N / 16;
  const int tn   = (int)(blockIdx.x % tiles_n);
  const int tm   = (int)(blockIdx.x / tiles_n) * 8 + (int)(threadIdx.x >> 5);
  const int lane = threadIdx.x & 31;
  const int half = lane >> 4;
  const int mr   = lane & 15;

  __shared__ unsigned short sh_w[16 * K];
#if ATHENA_HAS_TDM
  if (threadIdx.x < 32) {
    tdm_load_tile_2d((unsigned)(uintptr_t)&sh_w[0],
                     W + (size_t)tn * 16 * K, K, 16, K);
    __builtin_amdgcn_s_wait_tensorcnt(0);
  }
#else
  for (int i = threadIdx.x; i < 16 * K; i += blockDim.x)
    sh_w[i] = W[(size_t)tn * 16 * K + i];
#endif
  __syncthreads();

  if (tm * 16 >= M) return;

  const unsigned short* __restrict__ arow = A + (size_t)(tm * 16 + mr) * K;
  const int wbase = mr * K;

  v8f acc = {};
#pragma unroll
  for (int k0 = 0; k0 < K; k0 += 32) {
    if (k0 + 32 < K) __builtin_prefetch(arow + k0 + 32, 0, 1);
    v16us au, bu;
#pragma unroll
    for (int e = 0; e < 16; ++e) {
      const int ka = k0 + frag_k_a(e, half);
      const int kb = k0 + frag_k_b(e, half);
      au[e] = (ka < K) ? arow[ka] : (unsigned short)0;
      bu[e] = (kb < K) ? sh_w[wbase + kb] : (unsigned short)0;
    }
    v16bf a = __builtin_bit_cast(v16bf, au);
    v16bf b = __builtin_bit_cast(v16bf, bu);
    acc = __builtin_amdgcn_wmma_f32_16x16x32_bf16(
        false, a, false, b, (short)0, acc, false, false);
  }

  const int  col = tn * 16 + mr;
  const float bv = bias[col];
#pragma unroll
  for (int r = 0; r < 8; ++r) {
    const int row = tm * 16 + r + 8 * half;
    const int b   = row / PTS;
    float v = fmaxf(acc[r] + bv, 0.0f);
    atomicMax((int*)(out + (size_t)b * N + col), __float_as_int(v));
  }
}

// ---------------------------------------------------------------------------
// Confidence + concat: feat(B*N,4) = [sigmoid(x@cW^T+cb), x] in bf16
// ---------------------------------------------------------------------------
__global__ void conf_feat_kernel(const float* __restrict__ x,
                                 const float* __restrict__ cW,
                                 const float* __restrict__ cb,
                                 unsigned short* __restrict__ feat, int total)
{
  int t = blockIdx.x * blockDim.x + threadIdx.x;
  if (t >= total) return;
  float x0 = x[t * 3], x1 = x[t * 3 + 1], x2 = x[t * 3 + 2];
  float s = cW[0] * x0 + cW[1] * x1 + cW[2] * x2 + cb[0];
  float conf = 1.0f / (1.0f + __expf(-s));
  feat[t * 4 + 0] = f2bf(conf);
  feat[t * 4 + 1] = f2bf(x0);
  feat[t * 4 + 2] = f2bf(x1);
  feat[t * 4 + 3] = f2bf(x2);
}

// ---------------------------------------------------------------------------
// Farthest point sampling: one 512-thread workgroup per batch.
// Emits index BEFORE the update, matching the reference scan.
// ---------------------------------------------------------------------------
__global__ __launch_bounds__(512) void fps_kernel(
    const float* __restrict__ xyz, const int* __restrict__ init_far,
    int* __restrict__ out_idx, int N, int npoint)
{
  const int b   = blockIdx.x;
  const int tid = threadIdx.x;
  __shared__ float s_d[512];
  __shared__ int   s_i[512];
  __shared__ int   s_far;

  const float* __restrict__ xb = xyz + (size_t)b * N * 3;
  float px = 0.f, py = 0.f, pz = 0.f, dist = -INFINITY;
  if (tid < N) {
    px = xb[tid * 3]; py = xb[tid * 3 + 1]; pz = xb[tid * 3 + 2];
    dist = 1e10f;
  }
  if (tid == 0) s_far = init_far[b];
  __syncthreads();

  for (int it = 0; it < npoint; ++it) {
    const int far = s_far;
    if (tid == 0) out_idx[(size_t)b * npoint + it] = far;   // emit pre-update
    const float cx = xb[far * 3], cy = xb[far * 3 + 1], cz = xb[far * 3 + 2];
    if (tid < N) {
      const float dx = px - cx, dy = py - cy, dz = pz - cz;
      dist = fminf(dist, dx * dx + dy * dy + dz * dz);
    }
    s_d[tid] = dist;
    s_i[tid] = tid;
    __syncthreads();
    for (int s = 256; s > 0; s >>= 1) {
      if (tid < s) {
        const float da = s_d[tid], db = s_d[tid + s];
        const int   ia = s_i[tid], ib = s_i[tid + s];
        if (db > da || (db == da && ib < ia)) { s_d[tid] = db; s_i[tid] = ib; }
      }
      __syncthreads();
    }
    if (tid == 0) s_far = s_i[0];
    __syncthreads();
  }
}

// ---------------------------------------------------------------------------
// Gather + concat xyz: out(B,Ksel,C+3) = [src[b,idx,:C], bf16(xyz[b,idx,:])]
// ---------------------------------------------------------------------------
__global__ void gather_concat_kernel(
    const unsigned short* __restrict__ src, const float* __restrict__ xyz,
    const int* __restrict__ idx, unsigned short* __restrict__ out,
    int B, int Nsrc, int C, int Ksel, int Nxyz)
{
  long long t = (long long)blockIdx.x * blockDim.x + threadIdx.x;
  const int Ct = C + 3;
  const long long total = (long long)B * Ksel * Ct;
  if (t >= total) return;
  const int c = (int)(t % Ct);
  const long long rk = t / Ct;
  const int k = (int)(rk % Ksel);
  const int b = (int)(rk / Ksel);
  const int s = idx[(size_t)b * Ksel + k];
  unsigned short v;
  if (c < C) v = src[((size_t)b * Nsrc + s) * C + c];
  else       v = f2bf(xyz[((size_t)b * Nxyz + s) * 3 + (c - C)]);
  out[t] = v;
}

__global__ void f32_to_bf16_kernel(const float* __restrict__ in,
                                   unsigned short* __restrict__ out, int n)
{
  int t = blockIdx.x * blockDim.x + threadIdx.x;
  if (t < n) out[t] = f2bf(in[t]);
}

__global__ void zero_f32_kernel(float* __restrict__ p, int n)
{
  int t = blockIdx.x * blockDim.x + threadIdx.x;
  if (t < n) p[t] = 0.0f;
}

// ---------------------------------------------------------------------------
// Host orchestration
// ---------------------------------------------------------------------------
static inline int cdiv(long long a, long long b) { return (int)((a + b - 1) / b); }

template <int K, int N, int RELU>
static void launch_gemm(const unsigned short* A, const unsigned short* W,
                        const float* bias, unsigned short* out, int M,
                        hipStream_t stream)
{
  const int tiles_m = M / 16;                 // M is always a multiple of 128
  const int blocks  = (N / 16) * (tiles_m / 8);
  gemm_wmma<K, N, RELU><<<blocks, 256, 0, stream>>>(A, W, bias, out, M);
}

extern "C" void kernel_launch(void* const* d_in, const int* in_sizes, int n_in,
                              void* d_out, int out_size, void* d_ws, size_t ws_size,
                              hipStream_t stream)
{
  (void)in_sizes; (void)n_in; (void)out_size; (void)ws_size;
  constexpr int B = 128, N = 500;
  constexpr int M1 = B * N;        // 64000 rows for the first two stages
  constexpr int M2 = B * 1000;     // 128000 rows for the last stage

  const float* x      = (const float*)d_in[0];
  const float* conf_W = (const float*)d_in[1];
  const float* conf_b = (const float*)d_in[2];
  const float* w1 = (const float*)d_in[3];  const float* b1 = (const float*)d_in[4];
  const float* w2 = (const float*)d_in[5];  const float* b2 = (const float*)d_in[6];
  const float* pc1_W = (const float*)d_in[7]; const float* pc1_b = (const float*)d_in[8];
  const float* w3 = (const float*)d_in[9];  const float* b3 = (const float*)d_in[10];
  const float* w4 = (const float*)d_in[11]; const float* b4 = (const float*)d_in[12];
  const float* pc2_W = (const float*)d_in[13]; const float* pc2_b = (const float*)d_in[14];
  const float* w5 = (const float*)d_in[15]; const float* b5 = (const float*)d_in[16];
  const float* w6 = (const float*)d_in[17]; const float* b6 = (const float*)d_in[18];
  const int* init_far1 = (const int*)d_in[20];
  const int* init_far2 = (const int*)d_in[21];
  float* out = (float*)d_out;

  // Workspace carving
  char* base = (char*)d_ws;
  size_t off = 0;
  auto alloc = [&](size_t bytes) -> char* {
    char* p = base + off;
    off += (bytes + 255) & ~(size_t)255;
    return p;
  };
  auto bf = [&](size_t elems) -> unsigned short* {
    return (unsigned short*)alloc(elems * 2);
  };

  unsigned short* w1b  = bf(64 * 4);
  unsigned short* w2b  = bf(256 * 64);
  unsigned short* pc1b = bf(256 * 256);
  unsigned short* w3b  = bf(256 * 259);
  unsigned short* w4b  = bf(384 * 256);
  unsigned short* pc2b = bf(384 * 384);
  unsigned short* w5b  = bf(384 * 387);
  unsigned short* w6b  = bf(512 * 384);
  unsigned short* feat = bf((size_t)M1 * 4);
  unsigned short* h1   = bf((size_t)M1 * 64);
  unsigned short* l2   = bf((size_t)M1 * 256);
  unsigned short* l3   = bf((size_t)M1 * 256);
  int*            idx1 = (int*)alloc((size_t)M1 * 4);
  unsigned short* nf1  = bf((size_t)M1 * 259);
  unsigned short* h3   = bf((size_t)M1 * 256);
  unsigned short* l5   = bf((size_t)M1 * 384);
  unsigned short* l6   = bf((size_t)M1 * 384);
  int*            idx2 = (int*)alloc((size_t)M2 * 4);
  unsigned short* nf2  = bf((size_t)M2 * 387);
  unsigned short* h5   = bf((size_t)M2 * 384);

  // Weight conversion f32 -> bf16
  auto conv = [&](const float* src, unsigned short* dst, int n) {
    f32_to_bf16_kernel<<<cdiv(n, 256), 256, 0, stream>>>(src, dst, n);
  };
  conv(w1, w1b, 64 * 4);       conv(w2, w2b, 256 * 64);
  conv(pc1_W, pc1b, 256 * 256);conv(w3, w3b, 256 * 259);
  conv(w4, w4b, 384 * 256);    conv(pc2_W, pc2b, 384 * 384);
  conv(w5, w5b, 384 * 387);    conv(w6, w6b, 512 * 384);

  // Stage 0: confidence + concat
  conf_feat_kernel<<<cdiv(M1, 256), 256, 0, stream>>>(x, conf_W, conf_b, feat, M1);

  // Stage 1: feat(4) -> 64 -> 256, then pc1 (no act)
  launch_gemm<4, 64, 1>(feat, w1b, b1, h1, M1, stream);
  launch_gemm<64, 256, 1>(h1, w2b, b2, l2, M1, stream);
  launch_gemm<256, 256, 0>(l2, pc1b, pc1_b, l3, M1, stream);

  // FPS #1 and gather (l3 || x) -> nf1 (259)
  fps_kernel<<<B, 512, 0, stream>>>(x, init_far1, idx1, N, 500);
  {
    long long total = (long long)B * 500 * 259;
    gather_concat_kernel<<<cdiv(total, 256), 256, 0, stream>>>(
        l3, x, idx1, nf1, B, N, 256, 500, N);
  }

  // Stage 2: nf1(259) -> 256 -> 384, then pc2 (no act)
  launch_gemm<259, 256, 1>(nf1, w3b, b3, h3, M1, stream);
  launch_gemm<256, 384, 1>(h3, w4b, b4, l5, M1, stream);
  launch_gemm<384, 384, 0>(l5, pc2b, pc2_b, l6, M1, stream);

  // FPS #2 (1000 samples from 500 points) and gather (l6 || x) -> nf2 (387)
  fps_kernel<<<B, 512, 0, stream>>>(x, init_far2, idx2, N, 1000);
  {
    long long total = (long long)B * 1000 * 387;
    gather_concat_kernel<<<cdiv(total, 256), 256, 0, stream>>>(
        l6, x, idx2, nf2, B, N, 384, 1000, N);
  }

  // Stage 3: nf2(387) -> 384, then 384 -> 512 fused with relu + per-batch max
  launch_gemm<387, 384, 1>(nf2, w5b, b5, h5, M2, stream);
  zero_f32_kernel<<<cdiv(B * 512, 256), 256, 0, stream>>>(out, B * 512);
  {
    const int blocks = (512 / 16) * ((M2 / 16) / 8);
    gemm_wmma_maxout<384, 512, 1000><<<blocks, 256, 0, stream>>>(
        h5, w6b, b6, out, M2);
  }
}
